// FusedGATOp_16338055594701
// MI455X (gfx1250) — matrix operand order repaired and checked by reading, not compiled, not run on hardware
//
#include <hip/hip_runtime.h>

typedef __attribute__((ext_vector_type(16))) _Float16 v16h;
typedef __attribute__((ext_vector_type(8)))  float    v8f;

#define GAT_H   4
#define GAT_D   32
#define GAT_DEG 16
#define WAVES_PER_BLOCK 8

__device__ __forceinline__ float lrelu(float x) { return fmaxf(x, 0.2f * x); }

// One wave32 per destination node.
//  - lanes (l&15) = edge id for score/softmax phase (both halves redundant, no divergence)
//  - aggregation via V_WMMA_F32_16X16X32_F16:
//      K = 32 = {head h: edges 0..15} ++ {head h+1: edges 0..15}
//      A[m,K]  = X[c_j, h(K), dbase+m]   (features, f16, transposed gather)
//      B[K,n]  = (n<8 ? head h : head h+1 column mask) * alpha[h(K), j(K)]
//      D[m,n]  = out[h0, dbase+m] for n in 0..7, out[h1, dbase+m] for n in 8..15
//    -> 2 head-pairs x 2 d-chunks = 4 WMMAs per node, zero redundant feature loads.
__global__ __launch_bounds__(256, 4)
void gat_wmma_kernel(const float* __restrict__ attn_row,
                     const float* __restrict__ attn_col,
                     const float* __restrict__ in_feat,
                     const int*   __restrict__ row_indptr,
                     const int*   __restrict__ col_indices,
                     float* __restrict__ outp, int n)
{
    __shared__ float s_alpha[WAVES_PER_BLOCK][GAT_H * GAT_DEG]; // [h][j]
    __shared__ int   s_cidx [WAVES_PER_BLOCK][GAT_DEG];
    __shared__ float s_out  [WAVES_PER_BLOCK][GAT_H * GAT_D];   // 128 f32 per node

    const int lane = threadIdx.x & 31;
    const int wv   = threadIdx.x >> 5;
    const int node = blockIdx.x * WAVES_PER_BLOCK + wv;
    if (node >= n) return;                       // wave-uniform; EXEC stays full

    const int e    = lane & 15;                  // edge id (replicated in both halves)
    const int base = row_indptr[node];           // == node*16 for fixed-degree CSR

    // ---------- edge scores + numerically-stable softmax over 16 edges ----------
    const int    cj = col_indices[base + e];
    const float4 ac = *(const float4*)(attn_col + (size_t)cj   * GAT_H); // 16B gather
    const float4 ar = *(const float4*)(attn_row + (size_t)node * GAT_H); // uniform

    float sc[GAT_H] = { lrelu(ar.x + ac.x), lrelu(ar.y + ac.y),
                        lrelu(ar.z + ac.z), lrelu(ar.w + ac.w) };
    float mx[GAT_H], ex[GAT_H], sm[GAT_H];
    #pragma unroll
    for (int h = 0; h < GAT_H; ++h) mx[h] = sc[h];
    #pragma unroll
    for (int m = 1; m < 16; m <<= 1) {           // xor 1/2/4/8 stays inside 16-lane half
        #pragma unroll
        for (int h = 0; h < GAT_H; ++h) mx[h] = fmaxf(mx[h], __shfl_xor(mx[h], m));
    }
    #pragma unroll
    for (int h = 0; h < GAT_H; ++h) { ex[h] = __expf(sc[h] - mx[h]); sm[h] = ex[h]; }
    #pragma unroll
    for (int m = 1; m < 16; m <<= 1) {
        #pragma unroll
        for (int h = 0; h < GAT_H; ++h) sm[h] += __shfl_xor(sm[h], m);
    }
    if (lane < 16) {
        s_cidx[wv][e] = cj;
        #pragma unroll
        for (int h = 0; h < GAT_H; ++h) s_alpha[wv][h * GAT_DEG + e] = ex[h] / sm[h];
    }
    asm volatile("s_wait_dscnt 0" ::: "memory"); // intra-wave LDS visibility

    // ---------- WMMA aggregation ----------
    const int jb  = (lane >> 4) << 3;            // edge sub-block this lane feeds: 0 | 8
    const int m16 = lane & 15;                   // M index (feature dim within chunk)
    int cid[8];
    #pragma unroll
    for (int t = 0; t < 8; ++t) cid[t] = s_cidx[wv][jb + t];

    #pragma unroll
    for (int hp = 0; hp < 2; ++hp) {
        const int  h0   = hp * 2, h1 = hp * 2 + 1;
        const int  hsel = (lane < 16) ? h0 : h1;            // unused; each lane's B holds both h0 and h1
        const bool actB = ((lane & 15) < 8) == (lane < 16); // column-group mask
        const float* arow = &s_alpha[wv][hsel * GAT_DEG];

        // B[K,n]: same K order as A: elems 0..7 -> head h0 edge jb+k, elems 8..15 -> head h1 edge jb+k-8;
        // column n<8 takes head h0 only, n>=8 head h1 only.
        v16h B;
        #pragma unroll
        for (int k = 0; k < 16; ++k)
            B[k] = ((k < 8) == ((lane & 15) < 8)) ? (_Float16)s_alpha[wv][(k < 8 ? h0 : h1) * GAT_DEG + jb + (k & 7)] : (_Float16)0.0f;

        #pragma unroll
        for (int dc = 0; dc < 2; ++dc) {
            const int dbase = dc * 16;
            // A 16-bit 16x32 layout: lanes 0-15 elems 0..7 -> K0..7, 8..15 -> K16..23;
            // lanes 16-31 elems 0..7 -> K8..15, 8..15 -> K24..31.
            v16h A;
            #pragma unroll
            for (int t = 0; t < 8; ++t) {
                const float* fp = in_feat + (size_t)cid[t] * (GAT_H * GAT_D) + dbase + m16;
                A[t]     = (_Float16)fp[h0 * GAT_D];  // head h0, edge jb+t
                A[t + 8] = (_Float16)fp[h1 * GAT_D];  // head h1, edge jb+t
            }
            v8f c = {};
            c = __builtin_amdgcn_wmma_f32_16x16x32_f16(
                    /*neg_a=*/false, A, /*neg_b=*/false, B,
                    /*c_mod=*/(short)0, c, /*reuse_a=*/false, /*reuse_b=*/false);

            // D replicated across each column group; writer lanes 0,8,16,24.
            if ((lane & 7) == 0) {
                const int hh = hp * 2 + ((lane >> 3) & 1);        // column group -> head
                float* op = &s_out[wv][hh * GAT_D + dbase + ((lane >> 4) << 3)];
                float4 lo = { c[0], c[1], c[2], c[3] };
                float4 hi = { c[4], c[5], c[6], c[7] };
                *(float4*)op       = lo;
                *(float4*)(op + 4) = hi;
            }
        }
    }
    asm volatile("s_wait_dscnt 0" ::: "memory");

    // coalesced 512B store per wave
    const float4 r = *(const float4*)&s_out[wv][lane * 4];
    *(float4*)(outp + (size_t)node * (GAT_H * GAT_D) + lane * 4) = r;
}

extern "C" void kernel_launch(void* const* d_in, const int* in_sizes, int n_in,
                              void* d_out, int out_size, void* d_ws, size_t ws_size,
                              hipStream_t stream)
{
    const float* attn_row = (const float*)d_in[0];
    const float* attn_col = (const float*)d_in[1];
    const float* in_feat  = (const float*)d_in[2];
    const int*   indptr   = (const int*)d_in[3];
    const int*   colidx   = (const int*)d_in[4];
    float*       outp     = (float*)d_out;

    const int n = in_sizes[0] / GAT_H;           // N nodes (attn_row is N*H)
    dim3 block(256);                             // 8 waves -> 8 nodes per block
    dim3 grid((n + WAVES_PER_BLOCK - 1) / WAVES_PER_BLOCK);
    gat_wmma_kernel<<<grid, block, 0, stream>>>(attn_row, attn_col, in_feat,
                                                indptr, colidx, outp, n);
}
